// tPatchGNN_Classifier_43301860278557
// MI455X (gfx1250) — matrix operand order repaired and checked by reading, not compiled, not run on hardware
//
#include <hip/hip_runtime.h>
#include <hip/hip_bf16.h>
#include <math.h>

// ---------------------------------------------------------------------------
// tPatchGNN classifier forward for MI455X (gfx1250, wave32, WMMA).
// ~115 GFLOP of GEMMs on ~300MB of f16 activation traffic: f16 WMMA compute
// (~25us) balances HBM at 23.3 TB/s (~15-25us), so everything runs through
// v_wmma_f32_16x16x32_f16 with f32 accumulation.
// GEMM inner loop is fully branch-free:
//   - K always a multiple of 32; operands padded & 16B aligned
//   - A row pointers are CLAMPED (not guarded): OOB C rows are discarded at
//     store time, so clamped loads are safe and cost no exec-mask traffic
//   - each wave computes a 16(M)x64(N) tile: one A fragment (the streaming
//     67MB operand) feeds 4 WMMAs against 4 L2-resident B fragments
// V projections store transposed per 64-token batch so probs@V is A*B^T too.
// ---------------------------------------------------------------------------

typedef __attribute__((ext_vector_type(16))) _Float16 v16h;
typedef __attribute__((ext_vector_type(8)))  _Float16 v8h;
typedef __attribute__((ext_vector_type(8)))  float    v8f;

// ---------------- batched WMMA GEMM: C = act(alpha*A*B^T + bias)*rowScale ---
// B'(k,n) = B[z, n*ldb + k]; B rows must exist up to ceil64(N) (zero padded).
// transStore: C viewed as 64-row batches, stored transposed:
//   C[(m>>6)*Ncap*64 + n*64 + (m&63)]   (builds V^T for probs@V)
// n in [N, Ncap) stores 0 (keeps pad columns clean for downstream K=256).
__global__ void __launch_bounds__(128)
gemm_wmma_kernel(const _Float16* __restrict__ A, int lda, long long sA,
                 const _Float16* __restrict__ Bm, int ldb, long long sB,
                 void* Cv, int ldc, long long sC, int outF16, int Ncap,
                 int transStore,
                 const float* __restrict__ bias, const float* __restrict__ rowScale,
                 int M, int N, int K, float alpha, int relu)
{
    const int lane = threadIdx.x;                    // 0..31
    const int m0 = (blockIdx.x * 4 + threadIdx.y) * 16;  // wave-uniform
    if (m0 >= M) return;
    const int n0 = blockIdx.y * 64;
    const int z  = blockIdx.z;
    A  += (long long)z * sA;
    Bm += (long long)z * sB;

    const int hi = lane >> 4;        // 0: lanes 0-15, 1: lanes 16-31
    const int li = lane & 15;

    // A fragment: lane<16 -> row m0+li, K=k0+{0..7,16..23};
    //             lane>=16 -> K=k0+{8..15,24..31}. Row clamped to M-1.
    int ar = m0 + li; if (ar >= M) ar = M - 1;
    const _Float16* Ap = A + (size_t)ar * lda + (hi << 3);
    // B fragments for the 4 N sub-tiles:
    //   lane<16 -> col n0+t*16+li, K=k0+{0..7,16..23} ; lane>=16 -> K=k0+{8..15,24..31}
    const size_t bStep = (size_t)16 * ldb;
    const _Float16* Bp0 = Bm + (size_t)(n0 + li) * ldb + (hi << 3);
    const _Float16* Bp1 = Bp0 + bStep;
    const _Float16* Bp2 = Bp1 + bStep;
    const _Float16* Bp3 = Bp2 + bStep;

    v8f acc0 = {}, acc1 = {}, acc2 = {}, acc3 = {};

    auto loadA = [](const _Float16* p) -> v16h {
        v8h lo = *(const v8h*)p;
        v8h hv = *(const v8h*)(p + 16);
        v16h r;
#pragma unroll
        for (int i = 0; i < 8; ++i) { r[i] = lo[i]; r[8 + i] = hv[i]; }
        return r;
    };
    auto loadB = [](const _Float16* p) -> v16h {
        v8h lo = *(const v8h*)p;
        v8h hv = *(const v8h*)(p + 16);
        v16h r;
#pragma unroll
        for (int i = 0; i < 8; ++i) { r[i] = lo[i]; r[8 + i] = hv[i]; }
        return r;
    };

    for (int k0 = 0; k0 < K; k0 += 32) {
        __builtin_prefetch((const void*)(Ap + 64), 0, 1);   // streaming A
        v16h a  = loadA(Ap);
        v16h b0 = loadB(Bp0);
        acc0 = __builtin_amdgcn_wmma_f32_16x16x32_f16(false, a, false, b0, (short)0, acc0, false, false);
        v16h b1 = loadB(Bp1);
        acc1 = __builtin_amdgcn_wmma_f32_16x16x32_f16(false, a, false, b1, (short)0, acc1, false, false);
        v16h b2 = loadB(Bp2);
        acc2 = __builtin_amdgcn_wmma_f32_16x16x32_f16(false, a, false, b2, (short)0, acc2, false, false);
        v16h b3 = loadB(Bp3);
        acc3 = __builtin_amdgcn_wmma_f32_16x16x32_f16(false, a, false, b3, (short)0, acc3, false, false);
        Ap += 32; Bp0 += 32; Bp1 += 32; Bp2 += 32; Bp3 += 32;
    }

    // epilogue: C/D layout: VGPR r, lane<16 -> (M=r, N=li); lane>=16 -> (M=8+r)
    const int rb = hi * 8;
    auto epi = [&](const v8f& acc, int t) {
        const int n = n0 + t * 16 + li;
        const bool nOK = n < N;
        const float bv = (bias && nOK) ? bias[n] : 0.f;
#pragma unroll
        for (int r = 0; r < 8; ++r) {
            int m = m0 + rb + r;
            if (m >= M) continue;
            float val = acc[r] * alpha + bv;
            if (relu && val < 0.f) val = 0.f;
            if (rowScale) val *= rowScale[m];
            if (!nOK) {
                if (n >= Ncap) continue;
                val = 0.f;                     // zero-pad ragged column
            }
            size_t ci;
            if (transStore) ci = ((size_t)(m >> 6) * Ncap + n) * 64 + (m & 63);
            else            ci = (size_t)z * sC + (size_t)m * ldc + n;
            if (outF16) ((_Float16*)Cv)[ci] = (_Float16)val;
            else        ((float*)Cv)[ci]    = val;
        }
    };
    epi(acc0, 0); epi(acc1, 1); epi(acc2, 2); epi(acc3, 3);
}

// ---------------- softmax with optional bias / key-pad mask ----------------
// biasMode: 0 none; 1: += log(exp(-|t_q - t_j|*invTau)+1e-12) (ts per batch);
//           2: += log(exp(-|q-j|*delta*invTau)+1e-12)
__global__ void softmax_kernel(const float* __restrict__ S, int lds, long long sS,
                               _Float16* __restrict__ Pr, int ldp, long long sP,
                               int Ncol, int biasMode,
                               const float* __restrict__ ts, int sTs,
                               float invTau, float delta,
                               const float* __restrict__ mask, int sMask)
{
    const int z = blockIdx.z, qi = blockIdx.x, j = threadIdx.x;  // block=64
    __shared__ float sh[64];
    float s = -1e30f;
    if (j < Ncol) {
        s = S[(long long)z * sS + (long long)qi * lds + j];
        if (biasMode == 1) {
            float dt = fabsf(ts[(long long)z * sTs + qi] - ts[(long long)z * sTs + j]);
            s += logf(expf(-dt * invTau) + 1e-12f);
        } else if (biasMode == 2) {
            float dist = fabsf((float)(qi - j)) * delta;
            s += logf(expf(-dist * invTau) + 1e-12f);
        }
        if (mask && mask[(long long)z * sMask + j] <= 0.f) s = -1e9f;
    }
    sh[j] = s;
    __syncthreads();
    float mx = -1e30f;
    for (int i = 0; i < Ncol; ++i) mx = fmaxf(mx, sh[i]);
    float e = (j < Ncol) ? expf(s - mx) : 0.f;
    __syncthreads();
    sh[j] = e;
    __syncthreads();
    float sum = 0.f;
    for (int i = 0; i < Ncol; ++i) sum += sh[i];
    if (j < Ncol)
        Pr[(long long)z * sP + (long long)qi * ldp + j] = (_Float16)(e / sum);
}

// ---------------- layernorm over 256 channels, optional residual ------------
__global__ void ln_kernel(const _Float16* __restrict__ a, const _Float16* __restrict__ res,
                          const float* __restrict__ g, const float* __restrict__ bta,
                          _Float16* __restrict__ out)
{
    const int row = blockIdx.x, d = threadIdx.x;   // block=256
    __shared__ float sh[256];
    float v = (float)a[(size_t)row * 256 + d];
    if (res) v += (float)res[(size_t)row * 256 + d];
    sh[d] = v;
    __syncthreads();
    for (int s = 128; s > 0; s >>= 1) { if (d < s) sh[d] += sh[d + s]; __syncthreads(); }
    float mean = sh[0] / 256.f;
    __syncthreads();
    float c = v - mean;
    sh[d] = c * c;
    __syncthreads();
    for (int s = 128; s > 0; s >>= 1) { if (d < s) sh[d] += sh[d + s]; __syncthreads(); }
    float var = sh[0] / 256.f;
    out[(size_t)row * 256 + d] = (_Float16)(c * rsqrtf(var + 1e-5f) * g[d] + bta[d]);
}

// ---------------- x = [X_patch, te_scale(t), sin(te_periodic(t))], pad 32 ---
__global__ void build_x_kernel(const float* __restrict__ X, const float* __restrict__ ts,
                               const float* __restrict__ teSw, const float* __restrict__ teSb,
                               const float* __restrict__ tePw, const float* __restrict__ tePb,
                               _Float16* __restrict__ x, int R)
{
    int i = blockIdx.x * blockDim.x + threadIdx.x;
    if (i >= R) return;
    float t = ts[i];
    _Float16* o = x + (size_t)i * 32;
#pragma unroll
    for (int c = 0; c < 5; ++c) o[c] = (_Float16)X[i * 5 + c];
    o[5] = (_Float16)(t * teSw[0] + teSb[0]);
#pragma unroll
    for (int c = 0; c < 15; ++c) o[6 + c] = (_Float16)sinf(t * tePw[c] + tePb[c]);
#pragma unroll
    for (int c = 21; c < 32; ++c) o[c] = (_Float16)0.f;
}

// ---------------- intra readout-attention pooling ---------------------------
__global__ void intra_pool_kernel(const _Float16* __restrict__ outm,   // (2048,64,256)
                                  const _Float16* __restrict__ readout,
                                  const float* __restrict__ mask,
                                  _Float16* __restrict__ pooled,       // (2048,256)
                                  float scale)
{
    const int n = blockIdx.x, t = threadIdx.x;  // block=256
    __shared__ float sp[64];
    if (t < 64) {
        const _Float16* row = outm + ((size_t)n * 64 + t) * 256;
        float acc = 0.f;
        for (int d = 0; d < 255; ++d) acc += (float)readout[d] * (float)row[d];
        float s = acc * scale;
        if (mask[(size_t)n * 64 + t] <= 0.f) s = -1e9f;
        sp[t] = s;
    }
    __syncthreads();
    float mx = -1e30f;
    for (int j = 0; j < 64; ++j) mx = fmaxf(mx, sp[j]);
    float sum = 0.f;
    for (int j = 0; j < 64; ++j) sum += expf(sp[j] - mx);
    if (t < 255) {
        float acc = 0.f;
        for (int j = 0; j < 64; ++j)
            acc += expf(sp[j] - mx) * (float)outm[((size_t)n * 64 + j) * 256 + t];
        pooled[(size_t)n * 256 + t] = (_Float16)(acc / sum);
    } else if (t == 255) {
        pooled[(size_t)n * 256 + 255] = (_Float16)0.f;
    }
}

// ---------------- H = [pooled | cnt] + sinusoidal PE ------------------------
__global__ void assemble_H_kernel(const _Float16* __restrict__ pooled,
                                  const float* __restrict__ mask_pt,
                                  _Float16* __restrict__ H)
{
    int idx = blockIdx.x * blockDim.x + threadIdx.x;   // 2048*256
    if (idx >= 2048 * 256) return;
    int row = idx >> 8, d = idx & 255;
    int p = row & 63;
    float v;
    if (d < 255) v = (float)pooled[idx];
    else {
        float c = 0.f;
        for (int l = 0; l < 64; ++l) c += mask_pt[(size_t)row * 64 + l];
        v = c / 64.f;
    }
    float div = expf((float)(d & ~1) * (-logf(10000.f) / 256.f));
    float pe = (d & 1) ? cosf((float)p * div) : sinf((float)p * div);
    H[idx] = (_Float16)(v + pe);
}

// ---------------- final logits ---------------------------------------------
__global__ void logits_kernel(const _Float16* __restrict__ z, const float* __restrict__ w,
                              const float* __restrict__ b, float* __restrict__ out, int R)
{
    int r = blockIdx.x * blockDim.x + threadIdx.x;
    if (r >= R) return;
    float acc = 0.f;
    for (int d = 0; d < 256; ++d) acc += (float)z[(size_t)r * 256 + d] * w[d];
    out[r] = acc + b[0];
}

// ---------------- weight f32 -> f16, rows+cols zero padded ------------------
__global__ void cvt_pad_kernel(_Float16* __restrict__ dst, const float* __restrict__ src,
                               int N, int K, int ldd, int Nld)
{
    int i = blockIdx.x * blockDim.x + threadIdx.x;
    if (i >= Nld * ldd) return;
    int n = i / ldd, k = i % ldd;
    dst[i] = (n < N && k < K) ? (_Float16)src[(size_t)n * K + k] : (_Float16)0.f;
}

// ===========================================================================
extern "C" void kernel_launch(void* const* d_in, const int* in_sizes, int n_in,
                              void* d_out, int out_size, void* d_ws, size_t ws_size,
                              hipStream_t stream)
{
    (void)in_sizes; (void)n_in; (void)out_size; (void)ws_size;
    const int BP = 2048;          // B*P
    const int R  = 131072;        // B*P*L
    // d_in leaf order: setup_inputs() insertion order, params flattened in
    // insertion order with each linear as (W, b):
    //  0/1 te_scale W,b  2/3 te_periodic W,b  4/5 pp1  6/7 pp2
    //  8..15 intra q/k/v/o (W,b)  16 intra.readout
    //  17..28 enc0[0]: attn_in W,b attn_out W,b ff1 W,b ff2 W,b ln1_g,ln1_b,ln2_g,ln2_b
    //  29..36 gattn[0] q/k/v/o   37..44 gattn[1]
    //  45..56 enc[0][0]          57..68 enc[1][0]
    //  69 cls.queries 70/71 cls.attn_in 72/73 cls.attn_out 74/75 cls.logit
    //  76 X_patch  77 ts_patch  78 mask_pt  79 patch_mask
    auto F = [&](int i) { return (const float*)d_in[i]; };

    char* ws = (char*)d_ws;
    size_t off = 0;
    auto alloc = [&](size_t bytes) -> void* {
        void* p = ws + off;
        off = (off + bytes + 255) & ~(size_t)255;
        return p;
    };
    auto cvtW = [&](int idx, int N, int K, int ldd, int Nld) -> const _Float16* {
        _Float16* dst = (_Float16*)alloc((size_t)Nld * ldd * sizeof(_Float16));
        int n = Nld * ldd;
        cvt_pad_kernel<<<(n + 255) / 256, 256, 0, stream>>>(dst, F(idx), N, K, ldd, Nld);
        return dst;
    };
    auto gemm = [&](const _Float16* A, int lda, long long sA,
                    const _Float16* B, int ldb, long long sB,
                    void* C, int ldc, long long sC, int outF16, int Ncap, int tS,
                    const float* bias, const float* rowScale,
                    int M, int N, int K, float alpha, int relu, int nb) {
        dim3 grid((M + 63) / 64, (N + 63) / 64, nb);
        gemm_wmma_kernel<<<grid, dim3(32, 4), 0, stream>>>(
            A, lda, sA, B, ldb, sB, C, ldc, sC, outF16, Ncap, tS,
            bias, rowScale, M, N, K, alpha, relu);
    };

    // ---- weights -> f16 (cols padded for vector loads, rows to 64-multiples
    //      where used as B so every B-fragment load is in-bounds) ----
    const _Float16* w_pp1 = cvtW(4, 255, 21, 32, 256);
    const _Float16* w_pp2 = cvtW(6, 255, 255, 256, 256);
    const _Float16* w_iq  = cvtW(8,  255, 255, 256, 256);
    const _Float16* w_ik  = cvtW(10, 255, 255, 256, 256);
    const _Float16* w_iv  = cvtW(12, 255, 255, 256, 256);
    const _Float16* w_io  = cvtW(14, 255, 255, 256, 256);
    const _Float16* w_rd  = cvtW(16, 1, 255, 256, 1);
    struct EncW { const _Float16 *ain, *aout, *f1, *f2; int base; };
    const int encBase[3] = {17, 45, 57};
    EncW encw[3];
    for (int e = 0; e < 3; ++e) {
        int b = encBase[e];
        encw[e].ain  = cvtW(b + 0, 768, 256, 256, 768);
        encw[e].aout = cvtW(b + 2, 256, 256, 256, 256);
        encw[e].f1   = cvtW(b + 4, 2048, 256, 256, 2048);
        encw[e].f2   = cvtW(b + 6, 256, 2048, 2048, 256);
        encw[e].base = b;
    }
    const int gBase[2] = {29, 37};
    const _Float16* w_g[2][4];
    for (int i = 0; i < 2; ++i)
        for (int j = 0; j < 4; ++j)
            w_g[i][j] = cvtW(gBase[i] + 2 * j, 256, 256, 256, 256);
    const _Float16* w_cq  = cvtW(69, 5, 256, 256, 16);   // cls queries (A only)
    const _Float16* w_cin = cvtW(70, 768, 256, 256, 768);
    const _Float16* w_cao = cvtW(72, 256, 256, 256, 256);

    // ---- activation buffers (f16 except raw scores) ----
    _Float16* XB = (_Float16*)alloc((size_t)R * 32 * 2);
    _Float16* B0 = (_Float16*)alloc((size_t)R * 256 * 2);
    _Float16* B1 = (_Float16*)alloc((size_t)R * 256 * 2);
    _Float16* B2 = (_Float16*)alloc((size_t)R * 256 * 2);   // also VT (batch,256,64)
    _Float16* B3 = (_Float16*)alloc((size_t)R * 256 * 2);
    float*    SB = (float*)alloc((size_t)BP * 64 * 64 * 4);
    _Float16* PB = (_Float16*)alloc((size_t)BP * 64 * 64 * 2);
    _Float16* PL = (_Float16*)alloc((size_t)BP * 256 * 2);
    _Float16* H  = (_Float16*)alloc((size_t)BP * 256 * 2);
    _Float16* H2 = (_Float16*)alloc((size_t)BP * 256 * 2);
    _Float16* EQ = (_Float16*)alloc((size_t)BP * 256 * 2);
    _Float16* EK = (_Float16*)alloc((size_t)BP * 256 * 2);
    _Float16* EV = (_Float16*)alloc((size_t)BP * 256 * 2);  // V^T (32,256,64)
    _Float16* ECX= (_Float16*)alloc((size_t)BP * 256 * 2);
    _Float16* ET1= (_Float16*)alloc((size_t)BP * 256 * 2);
    _Float16* EFF= (_Float16*)alloc((size_t)BP * 2048 * 2);
    _Float16* CQ = (_Float16*)alloc((size_t)16 * 256 * 2);
    _Float16* CCX= (_Float16*)alloc((size_t)32 * 5 * 256 * 2);
    _Float16* ZB = (_Float16*)alloc((size_t)160 * 256 * 2);

    const float s255 = 1.f / sqrtf(255.f);
    const float s256 = 1.f / 16.f;
    const float invTau = 1.f / 300.f;

    // ---- patch pre-processing ----
    build_x_kernel<<<(R + 255) / 256, 256, 0, stream>>>(
        F(76), F(77), F(0), F(1), F(2), F(3), XB, R);
    gemm(XB, 32, 0, w_pp1, 32, 0, B0, 256, 0, 1, 256, 0, F(5), nullptr,
         R, 255, 32, 1.f, 1, 1);                                   // h1 = relu(pp1)
    gemm(B0, 256, 0, w_pp2, 256, 0, B1, 256, 0, 1, 256, 0, F(7), nullptr,
         R, 255, 256, 1.f, 0, 1);                                  // tok = pp2

    // ---- intra-patch attention (2048 batches of L=64) ----
    gemm(B1, 256, 0, w_iq, 256, 0, B3, 256, 0, 1, 256, 0, F(9),  nullptr, R, 255, 256, 1.f, 0, 1);
    gemm(B1, 256, 0, w_ik, 256, 0, B0, 256, 0, 1, 256, 0, F(11), nullptr, R, 255, 256, 1.f, 0, 1);
    gemm(B1, 256, 0, w_iv, 256, 0, B2, 0,   0, 1, 256, 1, F(13), nullptr, R, 255, 256, 1.f, 0, 1); // V^T
    gemm(B3, 256, 64 * 256, B0, 256, 64 * 256, SB, 64, 4096, 0, 64, 0,
         nullptr, nullptr, 64, 64, 256, s255, 0, BP);              // Q K^T
    softmax_kernel<<<dim3(64, 1, BP), 64, 0, stream>>>(
        SB, 64, 4096, PB, 64, 4096, 64, 1, F(77), 64, invTau, 0.f, F(78), 64);
    gemm(PB, 64, 4096, B2, 64, 256 * 64, B1, 256, 64 * 256, 1, 256, 0,
         nullptr, nullptr, 64, 255, 64, 1.f, 0, BP);               // P V
    gemm(B1, 256, 0, w_io, 256, 0, B0, 256, 0, 1, 256, 0, F(15), F(78),
         R, 255, 256, 1.f, 0, 1);                                  // o-proj * mask
    intra_pool_kernel<<<BP, 256, 0, stream>>>(B0, w_rd, F(78), PL, s255);
    assemble_H_kernel<<<(BP * 256 + 255) / 256, 256, 0, stream>>>(PL, F(78), H);

    // ---- transformer encoder layer ----
    auto encLayer = [&](const _Float16* Hin, _Float16* Hout, const EncW& w) {
        int b = w.base;
        gemm(Hin, 256, 0, w.ain,             256, 0, EQ, 256, 0, 1, 256, 0, F(b + 1),       nullptr, BP, 256, 256, 1.f, 0, 1);
        gemm(Hin, 256, 0, w.ain + 256 * 256, 256, 0, EK, 256, 0, 1, 256, 0, F(b + 1) + 256, nullptr, BP, 256, 256, 1.f, 0, 1);
        gemm(Hin, 256, 0, w.ain + 512 * 256, 256, 0, EV, 0,   0, 1, 256, 1, F(b + 1) + 512, nullptr, BP, 256, 256, 1.f, 0, 1);
        gemm(EQ, 256, 64 * 256, EK, 256, 64 * 256, SB, 64, 4096, 0, 64, 0,
             nullptr, nullptr, 64, 64, 256, s256, 0, 32);
        softmax_kernel<<<dim3(64, 1, 32), 64, 0, stream>>>(
            SB, 64, 4096, PB, 64, 4096, 64, 0, nullptr, 0, 0.f, 0.f, nullptr, 0);
        gemm(PB, 64, 4096, EV, 64, 256 * 64, ECX, 256, 64 * 256, 1, 256, 0,
             nullptr, nullptr, 64, 256, 64, 1.f, 0, 32);
        gemm(ECX, 256, 0, w.aout, 256, 0, ET1, 256, 0, 1, 256, 0, F(b + 3), nullptr, BP, 256, 256, 1.f, 0, 1);
        ln_kernel<<<BP, 256, 0, stream>>>(Hin, ET1, F(b + 8), F(b + 9), EQ);    // H1
        gemm(EQ, 256, 0, w.f1, 256, 0, EFF, 2048, 0, 1, 2048, 0, F(b + 5), nullptr, BP, 2048, 256, 1.f, 1, 1);
        gemm(EFF, 2048, 0, w.f2, 2048, 0, ET1, 256, 0, 1, 256, 0, F(b + 7), nullptr, BP, 256, 2048, 1.f, 0, 1);
        ln_kernel<<<BP, 256, 0, stream>>>(EQ, ET1, F(b + 10), F(b + 11), Hout);
    };
    // ---- graph attention layer ----
    auto gattnLayer = [&](const _Float16* Hin, _Float16* Hout, int gi) {
        int b = gBase[gi];
        gemm(Hin, 256, 0, w_g[gi][0], 256, 0, EQ, 256, 0, 1, 256, 0, F(b + 1), nullptr, BP, 256, 256, 1.f, 0, 1);
        gemm(Hin, 256, 0, w_g[gi][1], 256, 0, EK, 256, 0, 1, 256, 0, F(b + 3), nullptr, BP, 256, 256, 1.f, 0, 1);
        gemm(Hin, 256, 0, w_g[gi][2], 256, 0, EV, 0,   0, 1, 256, 1, F(b + 5), nullptr, BP, 256, 256, 1.f, 0, 1);
        gemm(EQ, 256, 64 * 256, EK, 256, 64 * 256, SB, 64, 4096, 0, 64, 0,
             nullptr, nullptr, 64, 64, 256, s256, 0, 32);
        softmax_kernel<<<dim3(64, 1, 32), 64, 0, stream>>>(
            SB, 64, 4096, PB, 64, 4096, 64, 2, nullptr, 0, invTau, 300.f, F(79), 64);
        gemm(PB, 64, 4096, EV, 64, 256 * 64, ECX, 256, 64 * 256, 1, 256, 0,
             nullptr, nullptr, 64, 256, 64, 1.f, 0, 32);
        gemm(ECX, 256, 0, w_g[gi][3], 256, 0, Hout, 256, 0, 1, 256, 0, F(b + 7), F(79), BP, 256, 256, 1.f, 0, 1);
    };

    encLayer(H, H2, encw[0]);
    gattnLayer(H2, H, 0);
    encLayer(H, H2, encw[1]);
    gattnLayer(H2, H, 1);
    encLayer(H, H2, encw[2]);

    // ---- classification head ----
    gemm(w_cq, 256, 0, w_cin,            256, 0, CQ, 256, 0, 1, 256, 0, F(71),       nullptr, 5,  256, 256, 1.f, 0, 1);
    gemm(H2, 256, 0,  w_cin + 256 * 256, 256, 0, EK, 256, 0, 1, 256, 0, F(71) + 256, nullptr, BP, 256, 256, 1.f, 0, 1);
    gemm(H2, 256, 0,  w_cin + 512 * 256, 256, 0, EV, 0,   0, 1, 256, 1, F(71) + 512, nullptr, BP, 256, 256, 1.f, 0, 1);
    gemm(CQ, 256, 0, EK, 256, 64 * 256, SB, 64, 320, 0, 64, 0,
         nullptr, nullptr, 5, 64, 256, s256, 0, 32);
    softmax_kernel<<<dim3(5, 1, 32), 64, 0, stream>>>(
        SB, 64, 320, PB, 64, 320, 64, 0, nullptr, 0, 0.f, 0.f, F(79), 64);
    gemm(PB, 64, 320, EV, 64, 256 * 64, CCX, 256, 5 * 256, 1, 256, 0,
         nullptr, nullptr, 5, 256, 64, 1.f, 0, 32);
    gemm(CCX, 256, 0, w_cao, 256, 0, ZB, 256, 0, 1, 256, 0, F(73), nullptr, 160, 256, 256, 1.f, 0, 1);
    logits_kernel<<<1, 256, 0, stream>>>(ZB, F(74), F(75), (float*)d_out, 160);
}
